// Sub_Mat_10015863734379
// MI455X (gfx1250) — matrix operand order repaired and checked by reading, not compiled, not run on hardware
//
#include <hip/hip_runtime.h>
#include <hip/hip_bf16.h>
#include <math.h>

typedef __attribute__((ext_vector_type(16))) _Float16 v16h;
typedef __attribute__((ext_vector_type(8)))  float    v8f;

// ---------------- problem constants ----------------
#define NN    4096
#define F_IN  500
#define HID   64
#define NC    7
#define NE    65536
#define NB    1024
#define NA    20
#define NCUT  32768        // E * CONF_R
#define WPR   128          // words per bitset row (4096/32)

// ---------------- workspace layout (u32 units) ----------------
#define OFF_HIST1  0u
#define OFF_HIST2  65536u
#define OFF_META   131072u            // meta[0]=hi-bin, meta[1]=residual rank, meta[2]=thresh bits
#define OFF_ADJ    131088u            // 4096*128 u32
#define OFF_HOP2   655376u            // 4096*128 u32
#define OFF_EMBED  1179664u           // 4096*64 f32
#define OFF_ANCH   1441808u           // 7*64 f32
#define OFF_CLS    1442256u           // 4096 i32
#define OFF_RED    1446352u           // 3*1024*448 f32
#define OFF_XH     2822608u           // 256 tiles * 16 chunks * 32 lanes * v16h = 4 MB
#define OFF_BH     3871184u           // 4 tiles * 16 chunks * 32 lanes * v16h = 64 KB
#define ZERO_WORDS 655376u            // hist1+hist2+meta+adj

// ---------------- utility: zero scratch ----------------
__global__ void k_zero(unsigned* __restrict__ p, unsigned n) {
    for (unsigned i = blockIdx.x * blockDim.x + threadIdx.x; i < n;
         i += gridDim.x * blockDim.x) p[i] = 0u;
}

// ---------------- radix-select for top-k threshold ----------------
__global__ void k_hist1(const float* __restrict__ w, unsigned* __restrict__ h) {
    int e = blockIdx.x * blockDim.x + threadIdx.x;
    if (e >= NE) return;
    unsigned bits = __float_as_uint(w[e]);
    atomicAdd(&h[bits >> 16], 1u);
}

__global__ void k_hist2(const float* __restrict__ w, const unsigned* __restrict__ meta,
                        unsigned* __restrict__ h) {
    int e = blockIdx.x * blockDim.x + threadIdx.x;
    if (e >= NE) return;
    unsigned bits = __float_as_uint(w[e]);
    if ((bits >> 16) == meta[0]) atomicAdd(&h[bits & 0xFFFFu], 1u);
}

// phase 0: find hi 16-bit bin holding the NCUT-th largest + residual rank
// phase 1: find lo 16 bits; write exact threshold bit pattern
__global__ __launch_bounds__(256) void k_select(const unsigned* __restrict__ hist,
                                                unsigned* __restrict__ meta, int phase) {
    __shared__ unsigned partial[256];
    int t = threadIdx.x;
    unsigned s = 0;
    for (int i = 0; i < 256; ++i) s += hist[t * 256 + i];
    partial[t] = s;
    __syncthreads();
    if (t == 0) {
        unsigned target = (phase == 0) ? (unsigned)NCUT : meta[1];
        unsigned cum = 0;
        int chunk = 0;
        for (int c = 255; c >= 0; --c) {
            if (cum + partial[c] >= target) { chunk = c; break; }
            cum += partial[c];
        }
        unsigned bin = 0, kp = target;
        for (int b = chunk * 256 + 255; b >= chunk * 256; --b) {
            unsigned hc = hist[b];
            if (cum + hc >= target) { bin = (unsigned)b; kp = target - cum; break; }
            cum += hc;
        }
        if (phase == 0) { meta[0] = bin; meta[1] = kp; }
        else           { meta[2] = (meta[0] << 16) | bin; }
    }
}

// ---------------- pruned adjacency as bitset ----------------
__global__ void k_adj(const int* __restrict__ ei, const float* __restrict__ w,
                      const unsigned* __restrict__ meta, unsigned* __restrict__ adj) {
    int e = blockIdx.x * blockDim.x + threadIdx.x;
    if (e >= NE) return;
    float th = __uint_as_float(meta[2]);
    if (w[e] >= th) {
        int r = ei[e], c = ei[NE + e];
        atomicOr(&adj[r * WPR + (c >> 5)], 1u << (c & 31));
    }
}

__global__ void k_diag(unsigned* __restrict__ adj) {
    int i = blockIdx.x * blockDim.x + threadIdx.x;
    if (i < NN) atomicOr(&adj[i * WPR + (i >> 5)], 1u << (i & 31));
}

// hop2 row i = OR of adj rows of all set bits in adj row i (diag makes it inclusive)
__global__ __launch_bounds__(128) void k_hop2(const unsigned* __restrict__ adj,
                                              unsigned* __restrict__ hop2) {
    int i = blockIdx.x, t = threadIdx.x;
    unsigned acc = 0u;
    for (int w = 0; w < WPR; ++w) {
        unsigned word = adj[i * WPR + w];
        while (word) {
            int b = __ffs(word) - 1;
            word &= word - 1;
            int k = w * 32 + b;
            acc |= adj[k * WPR + t];
        }
    }
    hop2[i * WPR + t] = acc;
}

// ---------------- encoder fragment pre-swizzle ----------------
// Materialize WMMA A fragments of x (f32 -> f16) in per-lane register order:
// frag index (tm*16+kk)*32 + lane, 16 contiguous halves per lane.
// K tail (500 -> 512) zero-padded here, once, so the GEMM loop is branch-free.
__global__ __launch_bounds__(32) void k_fraga(const float* __restrict__ x,
                                              _Float16* __restrict__ xh) {
    int tmkk = blockIdx.x;                 // 0..4095
    int tm = tmkk >> 4, kk = tmkk & 15;
    int l = threadIdx.x, m = l & 15, hi = l >> 4;
    v16h a;
#pragma unroll
    for (int j = 0; j < 16; ++j) {
        int ka = kk * 32 + hi * 16 + j;
        a[j] = (ka < F_IN) ? (_Float16)x[(tm * 16 + m) * F_IN + ka] : (_Float16)0.0f;
    }
    ((v16h*)xh)[tmkk * 32 + l] = a;
}

// Same for W_enc B fragments: frag index (tn*16+kk)*32 + lane.
__global__ __launch_bounds__(32) void k_fragb(const float* __restrict__ Wenc,
                                              _Float16* __restrict__ bh) {
    int tnkk = blockIdx.x;                 // 0..63
    int tn = tnkk >> 4, kk = tnkk & 15;
    int l = threadIdx.x, n = l & 15, hi = l >> 4;
    v16h b;
#pragma unroll
    for (int j = 0; j < 16; ++j) {
        int kb = kk * 32 + hi * 16 + j;
        b[j] = (kb < F_IN) ? (_Float16)Wenc[kb * HID + tn * 16 + n] : (_Float16)0.0f;
    }
    ((v16h*)bh)[tnkk * 32 + l] = b;
}

// ---------------- encoder GEMM via WMMA ----------------
// One wave per 16-row tile of embed; computes all 4 column tiles (16x64 out),
// reusing the A fragment 4x. Branch-free 16-iteration K loop:
// 5 coalesced 32B vector loads + 4 v_wmma per iteration.
__global__ __launch_bounds__(128) void k_encoder(const _Float16* __restrict__ xh,
                                                 const _Float16* __restrict__ bh,
                                                 const float* __restrict__ benc,
                                                 float* __restrict__ embed) {
    const int tm = blockIdx.x * 4 + (threadIdx.x >> 5);   // 0..255
    const int l  = threadIdx.x & 31;
    const v16h* Af = (const v16h*)xh + tm * 16 * 32;
    const v16h* Bf = (const v16h*)bh;
    v8f c[4] = {{}, {}, {}, {}};
    for (int kk = 0; kk < 16; ++kk) {
        v16h a = Af[kk * 32 + l];
        v16h b0 = Bf[(0 * 16 + kk) * 32 + l];
        v16h b1 = Bf[(1 * 16 + kk) * 32 + l];
        v16h b2 = Bf[(2 * 16 + kk) * 32 + l];
        v16h b3 = Bf[(3 * 16 + kk) * 32 + l];
        c[0] = __builtin_amdgcn_wmma_f32_16x16x32_f16(false, a, false, b0, (short)0, c[0], false, false);
        c[1] = __builtin_amdgcn_wmma_f32_16x16x32_f16(false, a, false, b1, (short)0, c[1], false, false);
        c[2] = __builtin_amdgcn_wmma_f32_16x16x32_f16(false, a, false, b2, (short)0, c[2], false, false);
        c[3] = __builtin_amdgcn_wmma_f32_16x16x32_f16(false, a, false, b3, (short)0, c[3], false, false);
    }
    const int m = l & 15, hi = l >> 4;
#pragma unroll
    for (int tn = 0; tn < 4; ++tn) {
        float bias = benc[tn * 16 + m];
#pragma unroll
        for (int r = 0; r < 8; ++r) {
            float v = c[tn][r] + bias;
            v = v > 0.f ? v : 0.f;
            embed[(tm * 16 + r + 8 * hi) * HID + tn * 16 + m] = v;
        }
    }
}

// ---------------- anchors, class assignment ----------------
__global__ void k_anchor(const float* __restrict__ embed, const int* __restrict__ aidx,
                         float* __restrict__ anchor) {
    int t = blockIdx.x * blockDim.x + threadIdx.x;
    if (t >= NC * HID) return;
    int c = t >> 6, h = t & 63;
    float s = 0.f;
    for (int a = 0; a < NA; ++a) s += embed[aidx[c * NA + a] * HID + h];
    anchor[t] = s * (1.0f / NA);
}

__global__ void k_cls(const float* __restrict__ embed, const float* __restrict__ anchor,
                      int* __restrict__ cls) {
    int i = blockIdx.x * blockDim.x + threadIdx.x;
    if (i >= NN) return;
    float best = 3.4e38f; int bc = 0;
    for (int c = 0; c < NC; ++c) {
        float d = 0.f;
        for (int h = 0; h < HID; ++h) {
            float df = embed[i * HID + h] - anchor[c * HID + h];
            d += df * df;
        }
        if (d < best) { best = d; bc = c; }
    }
    cls[i] = bc;
}

// ---------------- prediction head + log_softmax ----------------
__global__ void k_pred(const float* __restrict__ embed, const float* __restrict__ Wp,
                       const float* __restrict__ bp, float* __restrict__ out) {
    int i = blockIdx.x * blockDim.x + threadIdx.x;
    if (i >= NN) return;
    float v[NC];
    for (int c = 0; c < NC; ++c) {
        float s = bp[c];
        for (int h = 0; h < HID; ++h) s += embed[i * HID + h] * Wp[h * NC + c];
        v[c] = s;
    }
    float mx = v[0];
    for (int c = 1; c < NC; ++c) mx = fmaxf(mx, v[c]);
    float se = 0.f;
    for (int c = 0; c < NC; ++c) se += __expf(v[c] - mx);
    float lse = __logf(se);
    for (int c = 0; c < NC; ++c) out[i * NC + c] = v[c] - mx - lse;
}

// ---------------- class-mean over hop2 neighborhoods ----------------
// one 64-thread block per query row; thread h owns feature h, classes in VGPRs.
__global__ __launch_bounds__(64) void k_classmean(const unsigned* __restrict__ hop2,
                                                  const float* __restrict__ embed,
                                                  const int* __restrict__ cls,
                                                  const int* __restrict__ ego,
                                                  const int* __restrict__ pos,
                                                  const int* __restrict__ neg,
                                                  float* __restrict__ red) {
    int bid = blockIdx.x;                 // 0..3071
    int g = bid >> 10, b = bid & 1023;
    const int* src = (g == 0) ? ego : (g == 1) ? pos : neg;
    int q = src[b];
    int h = threadIdx.x;
    float acc[NC] = {}, cnt[NC] = {};
    for (int w = 0; w < WPR; ++w) {
        unsigned word = hop2[q * WPR + w];
        while (word) {
            int bpos = __ffs(word) - 1;
            word &= word - 1;
            int j = w * 32 + bpos;
            int cj = cls[j];
            float e = embed[j * HID + h];
#pragma unroll
            for (int c = 0; c < NC; ++c) {
                bool msk = (cj == c);
                acc[c] += msk ? e : 0.f;
                cnt[c] += msk ? 1.f : 0.f;
            }
        }
    }
    float* out = red + (size_t)bid * (NC * HID);
#pragma unroll
    for (int c = 0; c < NC; ++c) {
        float d = cnt[c] < 1.f ? 1.f : cnt[c];
        out[c * HID + h] = acc[c] / d;
    }
}

// ---------------- decoder: relu((e-o)^2 @ W1 + b1) @ W2 + b2 ----------------
__global__ __launch_bounds__(32) void k_dec(const float* __restrict__ red,
                                            const float* __restrict__ W1,
                                            const float* __restrict__ b1,
                                            const float* __restrict__ W2,
                                            const float* __restrict__ b2,
                                            float* __restrict__ out) {
    int g = blockIdx.x >> 10, b = blockIdx.x & 1023;   // g: 0=pos, 1=neg
    const float* e = red + (size_t)b * (NC * HID);
    const float* o = red + (size_t)(NB + g * NB + b) * (NC * HID);
    int k = threadIdx.x;                               // hidden unit 0..31
    float acc = b1[k];
    for (int j = 0; j < NC * HID; ++j) {
        float d = e[j] - o[j];
        acc += d * d * W1[j * 32 + k];
    }
    float hv = (acc > 0.f ? acc : 0.f) * W2[k];
    for (int off = 16; off; off >>= 1) hv += __shfl_xor(hv, off);
    if (k == 0) out[g * NB + b] = hv + b2[0];
}

// ---------------- driver ----------------
extern "C" void kernel_launch(void* const* d_in, const int* in_sizes, int n_in,
                              void* d_out, int out_size, void* d_ws, size_t ws_size,
                              hipStream_t stream) {
    const float* x    = (const float*)d_in[0];
    const int*   ei   = (const int*)  d_in[1];
    const int*   ego  = (const int*)  d_in[2];
    const int*   pos  = (const int*)  d_in[3];
    const int*   neg  = (const int*)  d_in[4];
    const float* ew   = (const float*)d_in[5];
    const int*   aidx = (const int*)  d_in[6];
    const float* Wenc = (const float*)d_in[8];
    const float* benc = (const float*)d_in[9];
    const float* Wp   = (const float*)d_in[10];
    const float* bp   = (const float*)d_in[11];
    const float* W1   = (const float*)d_in[12];
    const float* b1   = (const float*)d_in[13];
    const float* W2   = (const float*)d_in[14];
    const float* b2   = (const float*)d_in[15];

    unsigned*  ws    = (unsigned*)d_ws;
    unsigned*  hist1 = ws + OFF_HIST1;
    unsigned*  hist2 = ws + OFF_HIST2;
    unsigned*  meta  = ws + OFF_META;
    unsigned*  adj   = ws + OFF_ADJ;
    unsigned*  hop2  = ws + OFF_HOP2;
    float*     embed = (float*)(ws + OFF_EMBED);
    float*     anch  = (float*)(ws + OFF_ANCH);
    int*       cls   = (int*)  (ws + OFF_CLS);
    float*     red   = (float*)(ws + OFF_RED);
    _Float16*  xh    = (_Float16*)(ws + OFF_XH);
    _Float16*  bh    = (_Float16*)(ws + OFF_BH);
    float*     out   = (float*)d_out;     // [0,1024) pos, [1024,2048) neg, rest log_softmax

    k_zero   <<<1024, 256, 0, stream>>>(ws, ZERO_WORDS);
    k_hist1  <<<NE / 256, 256, 0, stream>>>(ew, hist1);
    k_select <<<1, 256, 0, stream>>>(hist1, meta, 0);
    k_hist2  <<<NE / 256, 256, 0, stream>>>(ew, meta, hist2);
    k_select <<<1, 256, 0, stream>>>(hist2, meta, 1);
    k_adj    <<<NE / 256, 256, 0, stream>>>(ei, ew, meta, adj);
    k_diag   <<<NN / 256, 256, 0, stream>>>(adj);
    k_fraga  <<<4096, 32, 0, stream>>>(x, xh);
    k_fragb  <<<64, 32, 0, stream>>>(Wenc, bh);
    k_encoder<<<64, 128, 0, stream>>>(xh, bh, benc, embed);
    k_hop2   <<<NN, 128, 0, stream>>>(adj, hop2);
    k_anchor <<<2, 256, 0, stream>>>(embed, aidx, anch);
    k_cls    <<<NN / 256, 256, 0, stream>>>(embed, anch, cls);
    k_pred   <<<NN / 256, 256, 0, stream>>>(embed, Wp, bp, out + 2 * NB);
    k_classmean<<<3 * NB, 64, 0, stream>>>(hop2, embed, cls, ego, pos, neg, red);
    k_dec    <<<2 * NB, 32, 0, stream>>>(red, W1, b1, W2, b2, out);
}
